// SequenceModel_82978768159580
// MI455X (gfx1250) — matrix operand order repaired and checked by reading, not compiled, not run on hardware
//
#include <hip/hip_runtime.h>
#include <math.h>

// ---------------- model constants ----------------
#define B_     2
#define N_     768
#define H_     4
#define DIM_   64
#define SK_    16
#define SV_    16
#define PK_    4
#define PV_    4
#define DEPTH_ 12
#define LN_EPS_ 1e-5f
#define EPS_    1e-8f
#define SCALAR_SCALE_ 0.17677669529663689f   /* (2*SK)^-0.5 = 1/sqrt(32) */
#define POINT_SCALE_  0.16666666666666666f   /* (2*PK*9/2)^-0.5 = 1/6    */

typedef _Float16 f16;
typedef __attribute__((ext_vector_type(16))) _Float16 v16h;
typedef __attribute__((ext_vector_type(8)))  _Float16 v8h;
typedef __attribute__((ext_vector_type(8)))  float    v8f;

// ---------------------------------------------------------------
// 16-lane row reductions as fused VOP2+DPP16 butterfly steps:
//   v_max_num_f32_dpp d, x, x row_ror:N  ==>  d = max(ror16(x, N), x)
// Steps 1,2,4,8 give every lane the full 16-lane reduction.
// One instruction per step, no LDS, no canonicalize overhead.
// ---------------------------------------------------------------
__device__ __forceinline__ float row16_max(float x)
{
    asm("v_max_num_f32_dpp %0, %0, %0 row_ror:1 row_mask:0xf bank_mask:0xf bound_ctrl:1" : "+v"(x));
    asm("v_max_num_f32_dpp %0, %0, %0 row_ror:2 row_mask:0xf bank_mask:0xf bound_ctrl:1" : "+v"(x));
    asm("v_max_num_f32_dpp %0, %0, %0 row_ror:4 row_mask:0xf bank_mask:0xf bound_ctrl:1" : "+v"(x));
    asm("v_max_num_f32_dpp %0, %0, %0 row_ror:8 row_mask:0xf bank_mask:0xf bound_ctrl:1" : "+v"(x));
    return x;
}
__device__ __forceinline__ float row16_sum(float x)
{
    asm("v_add_f32_dpp %0, %0, %0 row_ror:1 row_mask:0xf bank_mask:0xf bound_ctrl:1" : "+v"(x));
    asm("v_add_f32_dpp %0, %0, %0 row_ror:2 row_mask:0xf bank_mask:0xf bound_ctrl:1" : "+v"(x));
    asm("v_add_f32_dpp %0, %0, %0 row_ror:4 row_mask:0xf bank_mask:0xf bound_ctrl:1" : "+v"(x));
    asm("v_add_f32_dpp %0, %0, %0 row_ror:8 row_mask:0xf bank_mask:0xf bound_ctrl:1" : "+v"(x));
    return x;
}

// ---------------------------------------------------------------
// x[b,n,:] = single[b,n] * w_single[0,:] + b_single
// ---------------------------------------------------------------
__global__ void __launch_bounds__(64)
k_embed(const float* __restrict__ s, const float* __restrict__ w,
        const float* __restrict__ b, float* __restrict__ x)
{
    int row = blockIdx.x, t = threadIdx.x;
    x[(size_t)row * DIM_ + t] = s[row] * w[t] + b[t];
}

// ---------------------------------------------------------------
// Projections + point frame transform.  Writes WMMA-ready f16 tensors:
//   qc[bh][n][32]: d0..15 = q_s*SCALAR_SCALE, d16..27 = q_p_glob*pw*PS, pad 0
//   kc[bh][n][32]: d0..15 = k_s,              d16..27 = k_p_glob,       pad 0
//   vct[bh][c][n]: c0..15 = v_s,              c16..27 = v_p_glob,       pad 0
//   qb/kb[bh][n] = -0.5*pw*PS*{q,k}_sq   (rank-1 logit bias terms)
// ---------------------------------------------------------------
__global__ void __launch_bounds__(64)
k_proj(const float* __restrict__ x,
       const float* __restrict__ wq_s, const float* __restrict__ wk_s,
       const float* __restrict__ wv_s,
       const float* __restrict__ wq_p, const float* __restrict__ wk_p,
       const float* __restrict__ wv_p,
       const float* __restrict__ pweights,
       const float* __restrict__ rot, const float* __restrict__ trans,
       f16* __restrict__ qc, f16* __restrict__ kc, f16* __restrict__ vct,
       float* __restrict__ qb, float* __restrict__ kb)
{
    int row = blockIdx.x;                 // b*N + n
    int b = row / N_, n = row % N_;
    int t = threadIdx.x;                  // 0..63

    __shared__ float xr[64];
    __shared__ float lq[48], lk[48], lv[48];   // local-frame points
    __shared__ float gq[48], gk[48], gv[48];   // global-frame points
    __shared__ float sqs[64], sks[64], svs[64];

    xr[t] = x[(size_t)row * 64 + t];
    __syncthreads();

    float aq = 0.f, ak = 0.f, av = 0.f;
    #pragma unroll 8
    for (int j = 0; j < 64; ++j) {
        float xv = xr[j];
        aq += xv * wq_s[j * 64 + t];
        ak += xv * wk_s[j * 64 + t];
        av += xv * wv_s[j * 64 + t];
    }
    sqs[t] = aq; sks[t] = ak; svs[t] = av;

    if (t < 48) {
        float pq = 0.f, pk = 0.f, pv = 0.f;
        #pragma unroll 8
        for (int j = 0; j < 64; ++j) {
            float xv = xr[j];
            pq += xv * wq_p[j * 48 + t];
            pk += xv * wk_p[j * 48 + t];
            pv += xv * wv_p[j * 48 + t];
        }
        lq[t] = pq; lk[t] = pk; lv[t] = pv;
    }
    __syncthreads();

    if (t < 48) {
        // to_glob: glob[r] = sum_c loc[c]*R[c][r] + trans[r]  (einsum 'bncr')
        int hp = t / 3;                    // h*4 + pk
        int r  = t % 3;
        const float* R  = rot + (size_t)row * 9;
        float tr = trans[(size_t)row * 3 + r];
        float q_ = tr, k_ = tr, v_ = tr;
        #pragma unroll
        for (int c = 0; c < 3; ++c) {
            float Rc = R[c * 3 + r];
            q_ += lq[hp * 3 + c] * Rc;
            k_ += lk[hp * 3 + c] * Rc;
            v_ += lv[hp * 3 + c] * Rc;
        }
        gq[t] = q_; gk[t] = k_; gv[t] = v_;
    }
    __syncthreads();

    {   // scalar channels
        int h = t / 16, d = t % 16;
        size_t base = ((size_t)(b * H_ + h) * N_ + n) * 32;
        qc[base + d] = (f16)(sqs[t] * SCALAR_SCALE_);
        kc[base + d] = (f16)(sks[t]);
        vct[((size_t)(b * H_ + h) * 32 + d) * N_ + n] = (f16)(svs[t]);
    }
    if (t < 48) {   // point channels (fold pw*PS on the q side)
        int h2 = t / 12, d12 = t % 12;
        float pwps = log1pf(__expf(pweights[h2])) * POINT_SCALE_;
        size_t b2 = ((size_t)(b * H_ + h2) * N_ + n) * 32;
        qc[b2 + 16 + d12] = (f16)(gq[t] * pwps);
        kc[b2 + 16 + d12] = (f16)(gk[t]);
        vct[((size_t)(b * H_ + h2) * 32 + 16 + d12) * N_ + n] = (f16)(gv[t]);
    }
    if (t < 16) {   // zero padding channels 28..31
        int h3 = t / 4, j = t % 4;
        size_t b3 = ((size_t)(b * H_ + h3) * N_ + n) * 32;
        qc[b3 + 28 + j] = (f16)0.f;
        kc[b3 + 28 + j] = (f16)0.f;
        vct[((size_t)(b * H_ + h3) * 32 + 28 + j) * N_ + n] = (f16)0.f;
    }
    if (t < H_) {   // rank-1 bias terms
        float pwps = log1pf(__expf(pweights[t])) * POINT_SCALE_;
        float sq = 0.f, sk2 = 0.f;
        #pragma unroll
        for (int j = 0; j < 12; ++j) {
            float a = gq[t * 12 + j], c = gk[t * 12 + j];
            sq += a * a; sk2 += c * c;
        }
        size_t bi = (size_t)(b * H_ + t) * N_ + n;
        qb[bi] = -0.5f * pwps * sq;
        kb[bi] = -0.5f * pwps * sk2;
    }
}

// ---------------------------------------------------------------
// Flash-attention, one wave (32 lanes) per (b, h, 16-query tile).
// Per 32-key chunk: 2 logit WMMAs back-to-back, ONE fused online-softmax
// update (single-instruction DPP16 butterfly reductions), 2 P*V WMMAs.
// ---------------------------------------------------------------
__global__ void __launch_bounds__(32)
k_attn(const f16* __restrict__ qc, const f16* __restrict__ kc,
       const f16* __restrict__ vct,
       const float* __restrict__ qb, const float* __restrict__ kb,
       float* __restrict__ aout)
{
    const int blk  = blockIdx.x;
    const int qt   = blk % (N_ / 16);
    const int bh   = blk / (N_ / 16);
    const int lane = threadIdx.x;
    const int col  = lane & 15;
    const int hf   = lane >> 4;

    __shared__ f16 ptile[16 * 32];   // P staging: [q][k] f16

    // ---- A fragment: Qc tile (16 x 32 f16) ----
    v16h afrag;
    {
        const f16* qrow = qc + ((size_t)bh * N_ + qt * 16 + col) * 32;
        v8h lo = *(const v8h*)(qrow + hf * 16);       // K hf*16 .. +8
        v8h hi = *(const v8h*)(qrow + hf * 16 + 8);   // K hf*16+8 ..
        #pragma unroll
        for (int i = 0; i < 8; ++i) { afrag[i] = lo[i]; afrag[8 + i] = hi[i]; }
    }

    // per-lane q-bias for its 8 queries (q = qt*16 + r + 8*hf)
    float qbv[8];
    {
        const float* qbp = qb + (size_t)bh * N_ + qt * 16 + 8 * hf;
        #pragma unroll
        for (int r = 0; r < 8; ++r) qbv[r] = qbp[r];
    }

    float m8[8], l8[8];
    v8f acc0 = {}, acc1 = {};
    #pragma unroll
    for (int r = 0; r < 8; ++r) { m8[r] = -3.0e38f; l8[r] = 0.f; }

    const float* kbp   = kb + (size_t)bh * N_;
    const f16*   vbase = vct + (size_t)bh * 32 * N_;

    for (int kt = 0; kt < N_ / 32; ++kt) {
        const int k0 = kt * 32;

        // ---- two B fragments: Kc tiles (32 x 16 each) ----
        v16h bf0, bf1;
        {
            const f16* krow = kc + ((size_t)bh * N_ + k0 + col) * 32 + hf * 16;
            v8h lo = *(const v8h*)(krow);
            v8h hi = *(const v8h*)(krow + 8);
            #pragma unroll
            for (int i = 0; i < 8; ++i) { bf0[i] = lo[i]; bf0[8 + i] = hi[i]; }
        }
        {
            const f16* krow = kc + ((size_t)bh * N_ + k0 + 16 + col) * 32 + hf * 16;
            v8h lo = *(const v8h*)(krow);
            v8h hi = *(const v8h*)(krow + 8);
            #pragma unroll
            for (int i = 0; i < 8; ++i) { bf1[i] = lo[i]; bf1[8 + i] = hi[i]; }
        }
        const float kb0 = kbp[k0 + col];
        const float kb1 = kbp[k0 + 16 + col];
        v8f cb0, cb1;
        #pragma unroll
        for (int r = 0; r < 8; ++r) { cb0[r] = qbv[r] + kb0; cb1[r] = qbv[r] + kb1; }

        v8f s0 = __builtin_amdgcn_wmma_f32_16x16x32_f16(
                     false, afrag, false, bf0, (short)0, cb0, false, false);
        v8f s1 = __builtin_amdgcn_wmma_f32_16x16x32_f16(
                     false, afrag, false, bf1, (short)0, cb1, false, false);

        // ---- fused online softmax over the 32-key chunk ----
        float nm[8], p0[8], p1[8];
        #pragma unroll
        for (int r = 0; r < 8; ++r) {
            float tm = fmaxf(s0[r], s1[r]);          // across the two tiles
            tm = row16_max(tm);                      // across 16 key lanes
            nm[r] = fmaxf(m8[r], tm);
        }
        #pragma unroll
        for (int r = 0; r < 8; ++r) {
            p0[r] = __expf(s0[r] - nm[r]);
            p1[r] = __expf(s1[r] - nm[r]);
            float rs = row16_sum(p0[r] + p1[r]);
            float sc = __expf(m8[r] - nm[r]);
            l8[r] = l8[r] * sc + rs;
            m8[r] = nm[r];
            acc0[r] *= sc;
            acc1[r] *= sc;
        }
        // stage P (f16) to LDS in [q][k] order for A-fragment reload
        #pragma unroll
        for (int r = 0; r < 8; ++r) {
            ptile[(r + 8 * hf) * 32 + col]      = (f16)p0[r];
            ptile[(r + 8 * hf) * 32 + 16 + col] = (f16)p1[r];
        }
        __syncthreads();   // single-wave WG: lowers to dscnt wait

        // ---- A fragment: P tile (16 x 32 keys) from LDS ----
        v16h pfrag;
        {
            const f16* prow = ptile + col * 32;
            v8h lo = *(const v8h*)(prow + hf * 16);
            v8h hi = *(const v8h*)(prow + hf * 16 + 8);
            #pragma unroll
            for (int i = 0; i < 8; ++i) { pfrag[i] = lo[i]; pfrag[8 + i] = hi[i]; }
        }
        // ---- B fragments: Vct (32 keys x 16 channels), channels 0-15 / 16-31
        const int krow0 = k0 + hf * 16;
        {
            const f16* vr = vbase + (size_t)col * N_ + krow0;
            v16h bf; v8h lo = *(const v8h*)vr; v8h hi = *(const v8h*)(vr + 8);
            #pragma unroll
            for (int i = 0; i < 8; ++i) { bf[i] = lo[i]; bf[8 + i] = hi[i]; }
            acc0 = __builtin_amdgcn_wmma_f32_16x16x32_f16(
                        false, pfrag, false, bf, (short)0, acc0, false, false);
        }
        {
            const f16* vr = vbase + (size_t)(16 + col) * N_ + krow0;
            v16h bf; v8h lo = *(const v8h*)vr; v8h hi = *(const v8h*)(vr + 8);
            #pragma unroll
            for (int i = 0; i < 8; ++i) { bf[i] = lo[i]; bf[8 + i] = hi[i]; }
            acc1 = __builtin_amdgcn_wmma_f32_16x16x32_f16(
                        false, pfrag, false, bf, (short)0, acc1, false, false);
        }
        __syncthreads();   // protect ptile before next chunk overwrites
    }

    // ---- epilogue: out[q][c] = acc/l ----
    #pragma unroll
    for (int r = 0; r < 8; ++r) {
        int q = qt * 16 + r + 8 * hf;
        float inv = 1.0f / l8[r];
        float* orow = aout + ((size_t)bh * N_ + q) * 28;
        orow[col] = acc0[r] * inv;
        if (col < 12) orow[16 + col] = acc1[r] * inv;
    }
}

// ---------------------------------------------------------------
// Per-row epilogue: invert frames, norms, out-proj + residual + LN,
// 3-layer FFN + residual + LN.  Updates x in place (row-independent).
// ---------------------------------------------------------------
__global__ void __launch_bounds__(64)
k_post(const float* __restrict__ aout,
       const float* __restrict__ rot, const float* __restrict__ trans,
       const float* __restrict__ w_out, const float* __restrict__ b_out,
       const float* __restrict__ ang, const float* __restrict__ anb,
       const float* __restrict__ fw0, const float* __restrict__ fb0,
       const float* __restrict__ fw1, const float* __restrict__ fb1,
       const float* __restrict__ fw2, const float* __restrict__ fb2,
       const float* __restrict__ fng, const float* __restrict__ fnb,
       float* __restrict__ x)
{
    int row = blockIdx.x;
    int b = row / N_, n = row % N_;
    int t = threadIdx.x;

    __shared__ float res[128];
    __shared__ float sa[64], sb[64];

    {   // out_s -> res[0:64]
        int h = t / 16, d = t % 16;
        res[t] = aout[((size_t)(b * H_ + h) * N_ + n) * 28 + d];
    }
    if (t < 48) {   // out_p local frame: loc[r] = sum_c (glob[c]-t[c])*R[r][c]
        int hp = t / 3, r = t % 3;
        int h = hp / 4, pv = hp % 4;
        const float* R  = rot + (size_t)row * 9;
        const float* ao = aout + ((size_t)(b * H_ + h) * N_ + n) * 28 + 16;
        float acc = 0.f;
        #pragma unroll
        for (int c = 0; c < 3; ++c) {
            float g = ao[pv * 3 + c] - trans[(size_t)row * 3 + c];
            acc += g * R[r * 3 + c];
        }
        res[64 + t] = acc;
    }
    __syncthreads();
    if (t < 16) {   // point norms
        int off = 64 + t * 3;
        float a0 = res[off], a1 = res[off + 1], a2 = res[off + 2];
        res[112 + t] = sqrtf(a0 * a0 + a1 * a1 + a2 * a2 + EPS_);
    }
    __syncthreads();

    // out projection + residual
    float y = b_out[t];
    #pragma unroll 8
    for (int j = 0; j < 128; ++j) y += res[j] * w_out[j * 64 + t];
    y += x[(size_t)row * 64 + t];

    // layernorm 1
    sa[t] = y; __syncthreads();
    float mean = 0.f;
    for (int j = 0; j < 64; ++j) mean += sa[j];
    mean *= (1.f / 64.f);
    float var = 0.f;
    for (int j = 0; j < 64; ++j) { float d = sa[j] - mean; var += d * d; }
    var *= (1.f / 64.f);
    float xa = (y - mean) * rsqrtf(var + LN_EPS_) * ang[t] + anb[t];
    __syncthreads(); sa[t] = xa; __syncthreads();

    // FFN
    float h0 = fb0[t];
    #pragma unroll 8
    for (int j = 0; j < 64; ++j) h0 += sa[j] * fw0[j * 64 + t];
    h0 = fmaxf(h0, 0.f);
    sb[t] = h0; __syncthreads();
    float h1 = fb1[t];
    #pragma unroll 8
    for (int j = 0; j < 64; ++j) h1 += sb[j] * fw1[j * 64 + t];
    h1 = fmaxf(h1, 0.f);
    __syncthreads(); sb[t] = h1; __syncthreads();
    float h2 = fb2[t];
    #pragma unroll 8
    for (int j = 0; j < 64; ++j) h2 += sb[j] * fw2[j * 64 + t];
    float xn = h2 + xa;

    // layernorm 2
    __syncthreads(); sa[t] = xn; __syncthreads();
    mean = 0.f;
    for (int j = 0; j < 64; ++j) mean += sa[j];
    mean *= (1.f / 64.f);
    var = 0.f;
    for (int j = 0; j < 64; ++j) { float d = sa[j] - mean; var += d * d; }
    var *= (1.f / 64.f);
    x[(size_t)row * 64 + t] = (xn - mean) * rsqrtf(var + LN_EPS_) * fng[t] + fnb[t];
}

// ---------------------------------------------------------------
// Output head: out = x @ w_points + b_points   (20 outputs / row)
// ---------------------------------------------------------------
__global__ void __launch_bounds__(32)
k_points(const float* __restrict__ x, const float* __restrict__ wp,
         const float* __restrict__ bp, float* __restrict__ out)
{
    int row = blockIdx.x, t = threadIdx.x;
    __shared__ float xr[64];
    xr[t]      = x[(size_t)row * 64 + t];
    xr[t + 32] = x[(size_t)row * 64 + t + 32];
    __syncthreads();
    if (t < 20) {
        float a = bp[t];
        #pragma unroll 8
        for (int j = 0; j < 64; ++j) a += xr[j] * wp[j * 20 + t];
        out[(size_t)row * 20 + t] = a;
    }
}

// ---------------------------------------------------------------
extern "C" void kernel_launch(void* const* d_in, const int* in_sizes, int n_in,
                              void* d_out, int out_size, void* d_ws, size_t ws_size,
                              hipStream_t stream)
{
    (void)in_sizes; (void)n_in; (void)out_size; (void)ws_size;

    const float* single   = (const float*)d_in[0];
    /* d_in[1] = pair_repr, unused by reference */
    const float* rot      = (const float*)d_in[2];
    const float* trans    = (const float*)d_in[3];
    const float* w_single = (const float*)d_in[4];
    const float* b_single = (const float*)d_in[5];
    const float* wq_s     = (const float*)d_in[6];
    const float* wk_s     = (const float*)d_in[7];
    const float* wv_s     = (const float*)d_in[8];
    const float* wq_p     = (const float*)d_in[9];
    const float* wk_p     = (const float*)d_in[10];
    const float* wv_p     = (const float*)d_in[11];
    const float* pweights = (const float*)d_in[12];
    const float* w_out    = (const float*)d_in[13];
    const float* b_out    = (const float*)d_in[14];
    const float* ang      = (const float*)d_in[15];
    const float* anb      = (const float*)d_in[16];
    const float* fw0      = (const float*)d_in[17];
    const float* fb0      = (const float*)d_in[18];
    const float* fw1      = (const float*)d_in[19];
    const float* fb1      = (const float*)d_in[20];
    const float* fw2      = (const float*)d_in[21];
    const float* fb2      = (const float*)d_in[22];
    const float* fng      = (const float*)d_in[23];
    const float* fnb      = (const float*)d_in[24];
    const float* wp       = (const float*)d_in[25];
    const float* bp       = (const float*)d_in[26];

    char* ws = (char*)d_ws;
    float* x   = (float*)ws; ws += (size_t)B_ * N_ * DIM_ * 4;        // 384 KB
    f16*   qc  = (f16*)ws;   ws += (size_t)B_ * H_ * N_ * 32 * 2;     // 384 KB
    f16*   kc  = (f16*)ws;   ws += (size_t)B_ * H_ * N_ * 32 * 2;
    f16*   vct = (f16*)ws;   ws += (size_t)B_ * H_ * 32 * N_ * 2;
    float* qb  = (float*)ws; ws += (size_t)B_ * H_ * N_ * 4;          // 24 KB
    float* kb  = (float*)ws; ws += (size_t)B_ * H_ * N_ * 4;
    float* aout= (float*)ws; ws += (size_t)B_ * H_ * N_ * 28 * 4;     // 672 KB

    k_embed<<<B_ * N_, 64, 0, stream>>>(single, w_single, b_single, x);

    for (int layer = 0; layer < DEPTH_; ++layer) {
        k_proj<<<B_ * N_, 64, 0, stream>>>(x, wq_s, wk_s, wv_s,
                                           wq_p, wk_p, wv_p, pweights,
                                           rot, trans, qc, kc, vct, qb, kb);
        k_attn<<<B_ * H_ * (N_ / 16), 32, 0, stream>>>(qc, kc, vct, qb, kb, aout);
        k_post<<<B_ * N_, 64, 0, stream>>>(aout, rot, trans, w_out, b_out,
                                           ang, anb, fw0, fb0, fw1, fb1,
                                           fw2, fb2, fng, fnb, x);
    }

    k_points<<<B_ * N_, 32, 0, stream>>>(x, wp, bp, (float*)d_out);
}
